// LSTMAcceptor_15032385536011
// MI455X (gfx1250) — matrix operand order repaired and checked by reading, not compiled, not run on hardware
//
#include <hip/hip_runtime.h>
#include <hip/hip_bf16.h>

#define VOCAB 32000
#define EMB   256
#define HID   512
#define BATCH 256
#define SEQT  512
#define NGATE (4 * HID)            // 2048
#define KTOT  (EMB + HID)          // 768
#define KITERS (KTOT / 32)         // 24
#define KI_EMB (EMB / 32)          // 8  embedding-phase iterations
#define CTILES (NGATE / 16)        // 128 column tiles across all 4 gates
#define NTILES (HID / 16)          // 32 column tiles per gate
#define MTILES (BATCH / 16)        // 16 row tiles
#define FRAG_BYTES 1024            // 32 lanes * 16 bf16 * 2B

typedef __attribute__((ext_vector_type(16))) __bf16 v16bf;
typedef __attribute__((ext_vector_type(8)))  float  v8f;

struct alignas(32) bfpack { int4 lo; int4 hi; };

// ---- workspace layout (bytes), all offsets 256B aligned ----
#define WS_WB    0u                                                   // 3,145,728
#define WS_BIAS  (WS_WB + (unsigned)(KITERS * CTILES * FRAG_BYTES))   // + 8,192
#define WS_EMBBF (WS_BIAS + 2048u * 4u)                               // + 16,384,000
#define WS_HA    (WS_EMBBF + (unsigned)(VOCAB * EMB * 2))             // + 262,144 (bf16)
#define WS_HB    (WS_HA + (unsigned)(BATCH * HID * 2))                // + 262,144 (bf16)
#define WS_C     (WS_HB + (unsigned)(BATCH * HID * 2))                // + 524,288 (f32)

__device__ __forceinline__ __bf16 f2bf(float f) {
    unsigned u = __builtin_bit_cast(unsigned, f);
    unsigned r = (u + 0x7FFFu + ((u >> 16) & 1u)) >> 16;   // round-to-nearest-even
    unsigned short s = (unsigned short)r;
    return __builtin_bit_cast(__bf16, s);
}
__device__ __forceinline__ float bf2f(__bf16 b) {
    unsigned short s = __builtin_bit_cast(unsigned short, b);
    unsigned u = ((unsigned)s) << 16;
    return __builtin_bit_cast(float, u);
}
__device__ __forceinline__ float sigmoidf(float x) {
    return 1.0f / (1.0f + __expf(-x));
}

// ---------------------------------------------------------------------------
// Prologue 1: swizzle [W_ih | W_hh] into per-lane WMMA B-fragment order (bf16)
// and fold biases. Fragment (ki, ct): B[k][n], k = ki*32 + (lane>=16?16:0)+e,
// n = ct*16 + (lane&15); stored lane-contiguous (32B per lane).
// ---------------------------------------------------------------------------
__global__ void prep_weights(const float* __restrict__ W_ih,
                             const float* __restrict__ W_hh,
                             const float* __restrict__ b_ih,
                             const float* __restrict__ b_hh,
                             __bf16* __restrict__ wb,
                             float* __restrict__ bias) {
    int id = blockIdx.x * blockDim.x + threadIdx.x;   // 0 .. 1,572,864
    int e    = id & 15;
    int lane = (id >> 4) & 31;
    int frag = id >> 9;
    int ct   = frag & (CTILES - 1);
    int ki   = frag >> 7;

    int n = ct * 16 + (lane & 15);                    // global gate row (0..2047)
    int k = ki * 32 + ((lane >> 4) ? 16 : 0) + e;     // K index (0..767)
    float v = (k < EMB) ? W_ih[(size_t)n * EMB + k]
                        : W_hh[(size_t)n * HID + (k - EMB)];
    wb[id] = f2bf(v);

    if (id < NGATE) bias[id] = b_ih[id] + b_hh[id];
}

// Prologue 2: bf16 copy of the embedding table (time-invariant; L2-resident)
__global__ void prep_emb(const float* __restrict__ emb,
                         __bf16* __restrict__ emb_bf) {
    int id = blockIdx.x * blockDim.x + threadIdx.x;   // 0 .. 8,192,000
    emb_bf[id] = f2bf(emb[id]);
}

// zero h0 (bf16) and c0 (f32)
__global__ void zero_state(unsigned* __restrict__ hA_u32, float* __restrict__ c) {
    int id = blockIdx.x * blockDim.x + threadIdx.x;   // 0 .. 131072
    c[id] = 0.0f;
    if (id < (BATCH * HID / 2)) hA_u32[id] = 0u;      // 2 bf16 per u32
}

// ---------------------------------------------------------------------------
// One LSTM timestep: 512 waves; wave = (m_tile, n_tile) 16x16 output tile,
// 4 f32 accumulators (i,f,g,o). Fully unrolled K-loop in chunks of 2
// iterations: issue all 20 loads of the chunk (2 A frags + 8 B frags), then
// 8 v_wmma_f32_16x16x32_bf16 -> long load clauses + staggered partial waits.
// Fused cell update + length masking in the epilogue.
// ---------------------------------------------------------------------------
__global__ __launch_bounds__(256)
void lstm_step(const int* __restrict__ x,
               const int* __restrict__ lengths,
               const __bf16* __restrict__ emb_bf,
               const __bf16* __restrict__ wb,
               const float* __restrict__ bias,
               const __bf16* __restrict__ h_in,
               __bf16* __restrict__ h_out,
               float* __restrict__ c,
               int t) {
    const int lane   = threadIdx.x & 31;
    const int waveId = blockIdx.x * (blockDim.x >> 5) + (threadIdx.x >> 5);
    const int n_tile = waveId & (NTILES - 1);        // 0..31 (per-gate col tile)
    const int m_tile = waveId >> 5;                  // 0..15
    const int r16    = lane & 15;
    const int hi     = lane >> 4;                    // which K-half this lane holds
    const int halfsel = hi * 16;

    const int row = m_tile * 16 + r16;               // batch row this lane feeds
    const int idx = x[(size_t)row * SEQT + t];
    const char* abase_e = reinterpret_cast<const char*>(emb_bf + (size_t)idx * EMB + halfsel);
    const char* abase_h = reinterpret_cast<const char*>(h_in  + (size_t)row * HID + halfsel);
    const char* wbase   = reinterpret_cast<const char*>(wb) + (size_t)lane * 32;

    const int n = n_tile * 16 + r16;                 // hidden column this lane owns
    __builtin_prefetch(bias + n, 0, 1);              // warm epilogue bias (global_prefetch_b8)

    v8f acc[4] = {v8f{}, v8f{}, v8f{}, v8f{}};

    // A-fragment base for K-chunk ki (folds to a constant pointer after unroll)
#define A_BASE(ki_) (((ki_) < KI_EMB) ? (abase_e + (ki_) * 64)                 \
                                      : (abase_h + ((ki_) - KI_EMB) * 64))
#define B_PTR(ki_, g_)                                                         \
    reinterpret_cast<const v16bf*>(                                            \
        wbase + (size_t)((ki_) * CTILES + (g_) * NTILES + n_tile) * FRAG_BYTES)

    #pragma unroll
    for (int kc = 0; kc < KITERS; kc += 2) {
        // ---- issue all loads of this chunk first (one long clause) ----
        bfpack pa0, pa1;
        {
            const char* b0 = A_BASE(kc);
            pa0.lo = *reinterpret_cast<const int4*>(b0);
            pa0.hi = *reinterpret_cast<const int4*>(b0 + 16);
            const char* b1 = A_BASE(kc + 1);
            pa1.lo = *reinterpret_cast<const int4*>(b1);
            pa1.hi = *reinterpret_cast<const int4*>(b1 + 16);
        }
        v16bf B0[4], B1[4];
        #pragma unroll
        for (int g = 0; g < 4; ++g) B0[g] = *B_PTR(kc, g);
        #pragma unroll
        for (int g = 0; g < 4; ++g) B1[g] = *B_PTR(kc + 1, g);

        const v16bf A0 = __builtin_bit_cast(v16bf, pa0);
        const v16bf A1 = __builtin_bit_cast(v16bf, pa1);

        // ---- then the 8 WMMAs of the chunk ----
        #pragma unroll
        for (int g = 0; g < 4; ++g)
            acc[g] = __builtin_amdgcn_wmma_f32_16x16x32_bf16(
                false, A0, false, B0[g], (short)0, acc[g], false, false);
        #pragma unroll
        for (int g = 0; g < 4; ++g)
            acc[g] = __builtin_amdgcn_wmma_f32_16x16x32_bf16(
                false, A1, false, B1[g], (short)0, acc[g], false, false);
    }
#undef A_BASE
#undef B_PTR

    // ---- fused LSTM cell update ----
    const float bi = bias[n];
    const float bf = bias[HID + n];
    const float bg = bias[2 * HID + n];
    const float bo = bias[3 * HID + n];

    #pragma unroll
    for (int v = 0; v < 8; ++v) {
        const int orow = m_tile * 16 + v + hi * 8;   // C/D layout: lanes16-31 -> M+8
        const size_t p = (size_t)orow * HID + n;

        const float gi = sigmoidf(acc[0][v] + bi);
        const float gf = sigmoidf(acc[1][v] + bf);
        const float gg = tanhf(acc[2][v] + bg);
        const float go = sigmoidf(acc[3][v] + bo);

        const float cprev = c[p];
        const float hprev = bf2f(h_in[p]);
        float cn = gf * cprev + gi * gg;
        float hn = go * tanhf(cn);

        const bool live = t < lengths[orow];
        if (!live) { cn = cprev; hn = hprev; }
        c[p]     = cn;
        h_out[p] = f2bf(hn);
    }
}

// ---------------------------------------------------------------------------
// Head: out[b] = relu(h_T[b] @ W1^T + b1) @ W2^T + b2   (h_T is bf16)
// ---------------------------------------------------------------------------
__global__ void head_kernel(const __bf16* __restrict__ hT,
                            const float* __restrict__ W1,
                            const float* __restrict__ b1,
                            const float* __restrict__ W2,
                            const float* __restrict__ b2,
                            float* __restrict__ out) {
    const int b = blockIdx.x * blockDim.x + threadIdx.x;
    if (b >= BATCH) return;
    const __bf16* h = hT + (size_t)b * HID;
    float o = b2[0];
    #pragma unroll 1
    for (int j = 0; j < 16; ++j) {
        float a = b1[j];
        const float* w = W1 + (size_t)j * HID;
        for (int k = 0; k < HID; ++k) a += bf2f(h[k]) * w[k];
        o += fmaxf(a, 0.0f) * W2[j];
    }
    out[b] = o;
}

extern "C" void kernel_launch(void* const* d_in, const int* in_sizes, int n_in,
                              void* d_out, int out_size, void* d_ws, size_t ws_size,
                              hipStream_t stream) {
    const int*   x       = (const int*)  d_in[0];
    const int*   lengths = (const int*)  d_in[1];
    const float* emb     = (const float*)d_in[2];
    const float* W_ih    = (const float*)d_in[3];
    const float* W_hh    = (const float*)d_in[4];
    const float* b_ih    = (const float*)d_in[5];
    const float* b_hh    = (const float*)d_in[6];
    const float* W1      = (const float*)d_in[7];
    const float* b1      = (const float*)d_in[8];
    const float* W2      = (const float*)d_in[9];
    const float* b2      = (const float*)d_in[10];
    float* out = (float*)d_out;

    char* ws = (char*)d_ws;
    __bf16* wb     = (__bf16*)(ws + WS_WB);
    float*  bias   = (float*) (ws + WS_BIAS);
    __bf16* emb_bf = (__bf16*)(ws + WS_EMBBF);
    __bf16* hA     = (__bf16*)(ws + WS_HA);
    __bf16* hB     = (__bf16*)(ws + WS_HB);
    float*  cbuf   = (float*) (ws + WS_C);

    // 1) swizzle weights to bf16 B-fragments + fold biases
    prep_weights<<<(KITERS * CTILES * 512) / 256, 256, 0, stream>>>(
        W_ih, W_hh, b_ih, b_hh, wb, bias);
    // 2) bf16 embedding table (hoists all A-side conversions out of the loop)
    prep_emb<<<(VOCAB * EMB) / 256, 256, 0, stream>>>(emb, emb_bf);
    // 3) zero h0, c0
    zero_state<<<(BATCH * HID) / 256, 256, 0, stream>>>((unsigned*)hA, cbuf);

    // 4) 512 recurrent steps, ping-pong h (bf16)
    const int blocks = (MTILES * NTILES * 32) / 256; // 512 waves -> 64 blocks
    for (int t = 0; t < SEQT; ++t) {
        const __bf16* hin  = (t & 1) ? hB : hA;
        __bf16*       hout = (t & 1) ? hA : hB;
        lstm_step<<<blocks, 256, 0, stream>>>(x, lengths, emb_bf, wb, bias,
                                              hin, hout, cbuf, t);
    }
    // after t=511 (odd), h_T lives in hA
    // 5) MLP head
    head_kernel<<<1, 256, 0, stream>>>(hA, W1, b1, W2, b2, out);
}
